// Forward_33028298506349
// MI455X (gfx1250) — matrix operand order repaired and checked by reading, not compiled, not run on hardware
//
#include <hip/hip_runtime.h>
#include <hip/hip_bf16.h>

// CRF NLL for B=1024, T=512, K=48 on gfx1250.
//
// Kernel 1: sequence score (gather-sum), one wave per batch row -> d_ws.
// Kernel 2: forward algorithm. logsumexp recurrence rewritten as
//   alpha' = m + log( exp(alpha-m) @ exp(trans) ) + e_t
// so the K x K contraction runs on v_wmma_f32_16x16x32_f16 (K padded 48->64).
// Each wave owns a 16-batch tile for the whole T loop (64 waves total).

#define BB 1024
#define TT 512
#define KK 48
#define KPAD 64

typedef __attribute__((ext_vector_type(16))) _Float16 v16h;
typedef __attribute__((ext_vector_type(8)))  float    v8f;

// ---------------------------------------------------------------------------
// Kernel 1: unary + binary sequence score, one wave (32 lanes) per batch.
// ---------------------------------------------------------------------------
__global__ void crf_score_kernel(const float* __restrict__ emis,
                                 const int*   __restrict__ labels,
                                 const float* __restrict__ trans,
                                 float*       __restrict__ score_ws) {
    const int gtid = blockIdx.x * blockDim.x + threadIdx.x;
    const int b    = gtid >> 5;          // one wave per batch row
    const int lane = gtid & 31;
    if (b >= BB) return;

    float s = 0.0f;
    const int base = b * TT;
    for (int t = lane; t < TT; t += 32) {
        const int lab = labels[base + t];
        s += emis[(size_t)(base + t) * KK + lab];
        if (t > 0) {
            const int prev = labels[base + t - 1];
            s += trans[prev * KK + lab];
        }
    }
    // wave32 reduce
    #pragma unroll
    for (int mask = 16; mask >= 1; mask >>= 1)
        s += __shfl_xor(s, mask, 32);
    if (lane == 0) score_ws[b] = s;
}

// ---------------------------------------------------------------------------
// Kernel 2: forward algorithm with WMMA. 16 blocks x 128 threads = 64 waves,
// each wave handles 16 batch rows across all T steps.
// ---------------------------------------------------------------------------
__global__ void __launch_bounds__(128)
crf_forward_kernel(const float* __restrict__ emis,
                   const float* __restrict__ trans,
                   const float* __restrict__ score_ws,
                   float*       __restrict__ out) {
    const int tid  = threadIdx.x;
    const int wl   = tid >> 5;           // wave in block (0..3)
    const int lane = tid & 31;
    const int half = lane >> 4;          // 0: lanes 0-15, 1: lanes 16-31
    const int nIdx = lane & 15;
    const int b0   = (blockIdx.x * 4 + wl) * 16;   // first batch row of tile

    // Wave-private A staging buffer: 16 rows x KPAD halves (f16), row-major.
    __shared__ _Float16 ldsA[4][16][KPAD];

    // Zero (pads K=48..63 permanently; j loop never touches them again).
    for (int i = lane; i < 16 * KPAD; i += 32)
        (&ldsA[wl][0][0])[i] = (_Float16)0.0f;
    __syncthreads();

    // ---- B fragments: exp(trans) in f16, padded 48->64 on the K axis. ----
    // B layout (32x16 f16 per fragment): lane -> column N = nIdx (+16j tile),
    // halves v=0..15 -> K = kchunk*32 + half*16 + v.
    v16h Bf[2][3];
    #pragma unroll
    for (int kk = 0; kk < 2; ++kk)
        #pragma unroll
        for (int j = 0; j < 3; ++j)
            #pragma unroll
            for (int v = 0; v < 16; ++v) {
                const int Kr = kk * 32 + half * 16 + v;
                const int N  = j * 16 + nIdx;
                float val = 0.0f;
                if (Kr < KK) val = __expf(trans[Kr * KK + N]);
                Bf[kk][j][v] = (_Float16)val;
            }

    // ---- alpha0 = emissions[:, 0, :] loaded directly into C/D layout. ----
    // C/D 16x16 f32: VGPR r, lane -> M = r + 8*half (batch row), N = nIdx.
    v8f c[3];
    #pragma unroll
    for (int j = 0; j < 3; ++j)
        #pragma unroll
        for (int r = 0; r < 8; ++r) {
            const int row = b0 + r + 8 * half;
            c[j][r] = emis[(size_t)row * TT * KK + j * 16 + nIdx];
        }

    // ---- T-1 recurrence steps. ----
    for (int t = 1; t < TT; ++t) {
        // Row max over all 48 tags: per-lane max of the 3 tiles, then
        // shuffle-reduce within each 16-lane half (masks 1,2,4,8).
        float m[8];
        #pragma unroll
        for (int r = 0; r < 8; ++r) {
            float v = fmaxf(fmaxf(c[0][r], c[1][r]), c[2][r]);
            #pragma unroll
            for (int mask = 8; mask >= 1; mask >>= 1)
                v = fmaxf(v, __shfl_xor(v, mask, 32));
            m[r] = v;
        }

        // a = exp(alpha - m) -> LDS in row-major [batchRow][tag] f16.
        #pragma unroll
        for (int j = 0; j < 3; ++j)
            #pragma unroll
            for (int r = 0; r < 8; ++r)
                ldsA[wl][r + 8 * half][j * 16 + nIdx] =
                    (_Float16)__expf(c[j][r] - m[r]);
        __syncthreads();

        // A fragments (16x32 f16): lane row M = nIdx,
        // K = kchunk*32 + half*16 + v.
        v16h Af[2];
        #pragma unroll
        for (int kk = 0; kk < 2; ++kk)
            #pragma unroll
            for (int v = 0; v < 16; ++v) {
                const int Kr = kk * 32 + half * 16 + v;
                Af[kk][v] = ldsA[wl][nIdx][Kr];
            }

        // S = a @ exp(trans): 2 chained WMMAs per N-tile (K 0..31, 32..63).
        #pragma unroll
        for (int j = 0; j < 3; ++j) {
            v8f acc = {};
            acc = __builtin_amdgcn_wmma_f32_16x16x32_f16(
                false, Af[0], false, Bf[0][j], (short)0, acc, false, false);
            acc = __builtin_amdgcn_wmma_f32_16x16x32_f16(
                false, Af[1], false, Bf[1][j], (short)0, acc, false, false);

            // alpha' = m + log(S) + e_t
            #pragma unroll
            for (int r = 0; r < 8; ++r) {
                const int row = b0 + r + 8 * half;
                const float e = emis[(size_t)row * TT * KK +
                                     (size_t)t * KK + j * 16 + nIdx];
                c[j][r] = m[r] + __logf(acc[r]) + e;
            }
        }
        __syncthreads();
    }

    // ---- Final logsumexp over the 48 tags, then out = logZ - score. ----
    #pragma unroll
    for (int r = 0; r < 8; ++r) {
        float mv = fmaxf(fmaxf(c[0][r], c[1][r]), c[2][r]);
        #pragma unroll
        for (int mask = 8; mask >= 1; mask >>= 1)
            mv = fmaxf(mv, __shfl_xor(mv, mask, 32));
        float s = __expf(c[0][r] - mv) + __expf(c[1][r] - mv) +
                  __expf(c[2][r] - mv);
        #pragma unroll
        for (int mask = 8; mask >= 1; mask >>= 1)
            s += __shfl_xor(s, mask, 32);
        if (nIdx == 0) {
            const int row = b0 + r + 8 * half;
            out[row] = (mv + __logf(s)) - score_ws[row];
        }
    }
}

// ---------------------------------------------------------------------------
extern "C" void kernel_launch(void* const* d_in, const int* in_sizes, int n_in,
                              void* d_out, int out_size, void* d_ws, size_t ws_size,
                              hipStream_t stream) {
    const float* emis   = (const float*)d_in[0];   // (B, T, K) f32
    const int*   labels = (const int*)  d_in[1];   // (B, T)    i32
    const float* trans  = (const float*)d_in[2];   // (K, K)    f32
    float*       out    = (float*)d_out;           // (B,)      f32
    float*       score  = (float*)d_ws;            // B floats scratch

    // 1024 batches, one wave each: 128 blocks x 256 threads.
    crf_score_kernel<<<128, 256, 0, stream>>>(emis, labels, trans, score);
    // 64 batch-tiles of 16 rows: 16 blocks x 128 threads (4 waves/block).
    crf_forward_kernel<<<16, 128, 0, stream>>>(emis, trans, score, out);
}
